// ConsciousnessFiberAttention_12592844111902
// MI455X (gfx1250) — matrix operand order repaired and checked by reading, not compiled, not run on hardware
//
#include <hip/hip_runtime.h>
#include <hip/hip_bf16.h>
#include <math.h>

typedef __attribute__((ext_vector_type(16))) __bf16 v16bf;
typedef __attribute__((ext_vector_type(8)))  float  v8f;

union FragBF {
  v16bf v;
  uint4 u[2];
};

__device__ __forceinline__ unsigned short f32_to_bf16(float f) {
  unsigned int u = __float_as_uint(f);
  unsigned int rounded = u + 0x7FFFu + ((u >> 16) & 1u);   // round-to-nearest-even
  return (unsigned short)(rounded >> 16);
}

#define GFLAG_OUT_BF16 1
#define GFLAG_TRANSPOSE 2

// ------------------------------------------------------------------
// Shared epilogue for one 16x16 accumulator tile.
// ------------------------------------------------------------------
__device__ __forceinline__ void store_tile(
    v8f c, int row0, int col, void* Cv, int ldc, float alpha,
    const float* addMat, int ldAdd, const float* addRow, float addScale,
    const float* bias, int flags, int half)
{
  #pragma unroll
  for (int j = 0; j < 8; ++j) {
    const int row = row0 + half * 8 + j;
    float v = c[j] * alpha;
    if (addMat) v += addScale * (addMat[(size_t)row * ldAdd + col] + addRow[col]);
    if (bias)   v += bias[col];
    if (flags & GFLAG_OUT_BF16) {
      unsigned short hb = f32_to_bf16(v);
      unsigned short* Cb = (unsigned short*)Cv;
      if (flags & GFLAG_TRANSPOSE) Cb[(size_t)col * ldc + row] = hb;
      else                         Cb[(size_t)row * ldc + col] = hb;
    } else {
      ((float*)Cv)[(size_t)row * ldc + col] = v;
    }
  }
}

// ------------------------------------------------------------------
// Big-tile bf16 NT GEMM: C[M,N] = alpha*A[M,K]*B[N,K]^T (+extras)
// Block 256 threads = 8 waves (4 along M x 2 along N), wave tile 64x64
// (4x4 of 16x16 WMMA tiles -> 16 WMMAs per K-step, 32 FLOP/byte from L0).
// Requires M % 256 == 0, N % 128 == 0, K % 32 == 0.
// ------------------------------------------------------------------
__global__ __launch_bounds__(256) void gemm_bf16_nt_big(
    const unsigned short* __restrict__ A, int lda,
    const unsigned short* __restrict__ B, int ldb,
    void* __restrict__ Cv, int ldc, int K, float alpha,
    const float* __restrict__ addMat, int ldAdd,
    const float* __restrict__ addRow, float addScale,
    const float* __restrict__ bias, int flags)
{
  const int tid  = threadIdx.x;
  const int lane = tid & 31;
  const int wave = tid >> 5;
  const int wm   = wave & 3;   // 4 waves along M
  const int wn   = wave >> 2;  // 2 waves along N
  const int half = lane >> 4;
  const int r    = lane & 15;

  const int mBase = blockIdx.y * 256 + wm * 64;
  const int nBase = blockIdx.x * 128 + wn * 64;

  v8f acc[16] = {};

  const unsigned short* pa[4];
  const unsigned short* pb[4];
  #pragma unroll
  for (int i = 0; i < 4; ++i) {
    pa[i] = A + (size_t)(mBase + i * 16 + r) * lda;
    pb[i] = B + (size_t)(nBase + i * 16 + r) * ldb;
  }

  for (int k0 = 0; k0 < K; k0 += 32) {
    FragBF af[4], bfr[4];
    #pragma unroll
    for (int i = 0; i < 4; ++i) {
      af[i].u[0]  = *(const uint4*)(pa[i] + k0 + half * 8);
      af[i].u[1]  = *(const uint4*)(pa[i] + k0 + 16 + half * 8);
      bfr[i].u[0] = *(const uint4*)(pb[i] + k0 + half * 8);
      bfr[i].u[1] = *(const uint4*)(pb[i] + k0 + 16 + half * 8);
    }
    __builtin_prefetch(pa[0] + k0 + 128, 0, 1);
    __builtin_prefetch(pb[0] + k0 + 128, 0, 1);

    #pragma unroll
    for (int mt = 0; mt < 4; ++mt) {
      #pragma unroll
      for (int nt = 0; nt < 4; ++nt) {
        acc[mt * 4 + nt] = __builtin_amdgcn_wmma_f32_16x16x32_bf16(
            false, af[mt].v, false, bfr[nt].v, (short)0, acc[mt * 4 + nt], false, false);
      }
    }
  }

  #pragma unroll
  for (int mt = 0; mt < 4; ++mt) {
    #pragma unroll
    for (int nt = 0; nt < 4; ++nt) {
      store_tile(acc[mt * 4 + nt], mBase + mt * 16, nBase + nt * 16 + r,
                 Cv, ldc, alpha, addMat, ldAdd, addRow, addScale, bias, flags, half);
    }
  }
}

// ------------------------------------------------------------------
// Small-tile variant (block 128x64, wave tile 32x32) for narrow N.
// Requires M % 128 == 0, N % 64 == 0, K % 32 == 0.
// ------------------------------------------------------------------
__global__ __launch_bounds__(256) void gemm_bf16_nt(
    const unsigned short* __restrict__ A, int lda,
    const unsigned short* __restrict__ B, int ldb,
    void* __restrict__ Cv, int ldc, int K, float alpha,
    const float* __restrict__ addMat, int ldAdd,
    const float* __restrict__ addRow, float addScale,
    const float* __restrict__ bias, int flags)
{
  const int tid  = threadIdx.x;
  const int lane = tid & 31;
  const int wave = tid >> 5;
  const int wm   = wave & 3;
  const int wn   = wave >> 2;
  const int half = lane >> 4;
  const int r    = lane & 15;

  const int mBase = blockIdx.y * 128 + wm * 32;
  const int nBase = blockIdx.x * 64  + wn * 32;

  v8f acc[4] = {};

  const unsigned short* pa0 = A + (size_t)(mBase + r)      * lda;
  const unsigned short* pa1 = A + (size_t)(mBase + 16 + r) * lda;
  const unsigned short* pb0 = B + (size_t)(nBase + r)      * ldb;
  const unsigned short* pb1 = B + (size_t)(nBase + 16 + r) * ldb;

  for (int k0 = 0; k0 < K; k0 += 32) {
    FragBF a0, a1, b0, b1;
    a0.u[0] = *(const uint4*)(pa0 + k0 + half * 8);
    a0.u[1] = *(const uint4*)(pa0 + k0 + 16 + half * 8);
    a1.u[0] = *(const uint4*)(pa1 + k0 + half * 8);
    a1.u[1] = *(const uint4*)(pa1 + k0 + 16 + half * 8);
    b0.u[0] = *(const uint4*)(pb0 + k0 + half * 8);
    b0.u[1] = *(const uint4*)(pb0 + k0 + 16 + half * 8);
    b1.u[0] = *(const uint4*)(pb1 + k0 + half * 8);
    b1.u[1] = *(const uint4*)(pb1 + k0 + 16 + half * 8);

    __builtin_prefetch(pa0 + k0 + 128, 0, 1);
    __builtin_prefetch(pb0 + k0 + 128, 0, 1);

    acc[0] = __builtin_amdgcn_wmma_f32_16x16x32_bf16(false, a0.v, false, b0.v, (short)0, acc[0], false, false);
    acc[1] = __builtin_amdgcn_wmma_f32_16x16x32_bf16(false, a0.v, false, b1.v, (short)0, acc[1], false, false);
    acc[2] = __builtin_amdgcn_wmma_f32_16x16x32_bf16(false, a1.v, false, b0.v, (short)0, acc[2], false, false);
    acc[3] = __builtin_amdgcn_wmma_f32_16x16x32_bf16(false, a1.v, false, b1.v, (short)0, acc[3], false, false);
  }

  #pragma unroll
  for (int mt = 0; mt < 2; ++mt) {
    #pragma unroll
    for (int nt = 0; nt < 2; ++nt) {
      store_tile(acc[mt * 2 + nt], mBase + mt * 16, nBase + nt * 16 + r,
                 Cv, ldc, alpha, addMat, ldAdd, addRow, addScale, bias, flags, half);
    }
  }
}

// ------------------------------------------------------------------ helpers
__global__ __launch_bounds__(256) void cvt_f32_bf16_kernel(
    const float* __restrict__ in, unsigned short* __restrict__ out, int n) {
  int i = blockIdx.x * 256 + threadIdx.x;
  if (i < n) out[i] = f32_to_bf16(in[i]);
}

__global__ __launch_bounds__(256) void col_mean_kernel(
    const float* __restrict__ fp, float* __restrict__ mi, int rows, int cols) {
  __shared__ float red[256];
  int c = blockIdx.x, tid = threadIdx.x;
  float s = 0.f;
  for (int r2 = tid; r2 < rows; r2 += 256) s += fp[(size_t)r2 * cols + c];
  red[tid] = s; __syncthreads();
  for (int off = 128; off > 0; off >>= 1) { if (tid < off) red[tid] += red[tid + off]; __syncthreads(); }
  if (tid == 0) mi[c] = red[0] / (float)rows;
}

__global__ void fiber_mlp_kernel(const float* __restrict__ mi,
                                 const float* __restrict__ w1, const float* __restrict__ b1,
                                 const float* __restrict__ w2, const float* __restrict__ b2,
                                 float* __restrict__ raw) {
  __shared__ float smi[64];
  __shared__ float sh[32];
  int t = threadIdx.x;  // 64 threads
  smi[t] = mi[t];
  __syncthreads();
  if (t < 32) {
    float a = b1[t];
    #pragma unroll
    for (int j = 0; j < 64; ++j) a += w1[t * 64 + j] * smi[j];
    sh[t] = fmaxf(a, 0.f);
  }
  __syncthreads();
  float a = b2[t];
  #pragma unroll
  for (int j = 0; j < 32; ++j) a += w2[t * 32 + j] * sh[j];
  raw[t] = a;
}

__global__ __launch_bounds__(256) void metric_kernel(
    const float* __restrict__ raw, float* __restrict__ metric) {
  int idx = blockIdx.x * 256 + threadIdx.x;  // 4096 total
  int i = idx >> 6, j = idx & 63;
  float v = raw[i] * raw[j];
  if (i == j) v += 0.1f;
  metric[idx] = v;
}

// g = fp @ metric  (2048x64 * 64x64), written directly as bf16
__global__ void fiber_g_kernel(const float* __restrict__ fp, const float* __restrict__ metric,
                               unsigned short* __restrict__ gb) {
  __shared__ float smet[4096];
  int t = threadIdx.x;  // 64 threads
  for (int i = t; i < 4096; i += 64) smet[i] = metric[i];
  __syncthreads();
  for (int rr = 0; rr < 32; ++rr) {
    int s = blockIdx.x * 32 + rr;
    float a = 0.f;
    #pragma unroll
    for (int j = 0; j < 64; ++j) a += fp[(size_t)s * 64 + j] * smet[j * 64 + t];
    gb[(size_t)s * 64 + t] = f32_to_bf16(a);
  }
}

__global__ __launch_bounds__(256) void curv_kernel(
    const float* __restrict__ fp, const float* __restrict__ cm, float* __restrict__ curv) {
  int s = blockIdx.x * 256 + threadIdx.x;
  float a = 0.f;
  #pragma unroll
  for (int f = 0; f < 64; ++f) a += fp[(size_t)s * 64 + f] * cm[f * 65];  // diagonal
  curv[s] = a;
}

__global__ __launch_bounds__(256) void softmax_rows(
    const float* __restrict__ S, unsigned short* __restrict__ P, int n) {
  __shared__ float red[256];
  const int tid = threadIdx.x;
  const float* row = S + (size_t)blockIdx.x * n;
  unsigned short* out = P + (size_t)blockIdx.x * n;

  float m = -3.4e38f;
  for (int j = tid; j < n; j += 256) m = fmaxf(m, row[j]);
  red[tid] = m; __syncthreads();
  for (int off = 128; off > 0; off >>= 1) { if (tid < off) red[tid] = fmaxf(red[tid], red[tid + off]); __syncthreads(); }
  m = red[0]; __syncthreads();

  float s = 0.f;
  for (int j = tid; j < n; j += 256) s += __expf(row[j] - m);
  red[tid] = s; __syncthreads();
  for (int off = 128; off > 0; off >>= 1) { if (tid < off) red[tid] += red[tid + off]; __syncthreads(); }
  const float inv = 1.f / red[0];

  for (int j = tid; j < n; j += 256) out[j] = f32_to_bf16(__expf(row[j] - m) * inv);
}

__global__ __launch_bounds__(256) void reduce_rows_kernel(
    const float* __restrict__ M, float* __restrict__ partials, int n) {
  __shared__ float red[256];
  const float* row = M + (size_t)blockIdx.x * n;
  int tid = threadIdx.x;
  float s = 0.f;
  for (int j = tid; j < n; j += 256) s += row[j];
  red[tid] = s; __syncthreads();
  for (int off = 128; off > 0; off >>= 1) { if (tid < off) red[tid] += red[tid + off]; __syncthreads(); }
  if (tid == 0) partials[blockIdx.x] = red[0];
}

__global__ __launch_bounds__(256) void reduce_final_kernel(
    const float* __restrict__ partials, const float* __restrict__ curv,
    float* __restrict__ outScalar, int n) {
  __shared__ float red[256];
  int tid = threadIdx.x;
  float s1 = 0.f, s2 = 0.f;
  for (int j = tid; j < n; j += 256) { s1 += partials[j]; s2 += curv[j]; }
  red[tid] = s1; __syncthreads();
  for (int off = 128; off > 0; off >>= 1) { if (tid < off) red[tid] += red[tid + off]; __syncthreads(); }
  s1 = red[0]; __syncthreads();
  red[tid] = s2; __syncthreads();
  for (int off = 128; off > 0; off >>= 1) { if (tid < off) red[tid] += red[tid + off]; __syncthreads(); }
  if (tid == 0) *outScalar = s1 / ((float)n * (float)n) + red[0] / (float)n;
}

// ------------------------------------------------------------------ host
static void launch_gemm(hipStream_t st,
                        const unsigned short* A, int lda,
                        const unsigned short* B, int ldb,
                        void* C, int ldc, int M, int N, int K, float alpha,
                        const float* addMat, int ldAdd, const float* addRow, float addScale,
                        const float* bias, int flags) {
  if ((M % 256 == 0) && (N % 128 == 0)) {
    dim3 grid(N / 128, M / 256);
    gemm_bf16_nt_big<<<grid, 256, 0, st>>>(A, lda, B, ldb, C, ldc, K, alpha,
                                           addMat, ldAdd, addRow, addScale, bias, flags);
  } else {
    dim3 grid(N / 64, M / 128);
    gemm_bf16_nt<<<grid, 256, 0, st>>>(A, lda, B, ldb, C, ldc, K, alpha,
                                       addMat, ldAdd, addRow, addScale, bias, flags);
  }
}

static void launch_cvt(hipStream_t st, const float* in, unsigned short* out, int n) {
  cvt_f32_bf16_kernel<<<(n + 255) / 256, 256, 0, st>>>(in, out, n);
}

extern "C" void kernel_launch(void* const* d_in, const int* in_sizes, int n_in,
                              void* d_out, int out_size, void* d_ws, size_t ws_size,
                              hipStream_t stream) {
  (void)in_sizes; (void)n_in; (void)out_size; (void)ws_size;
  constexpr int S = 2048, D = 2048, F = 64, H = 16, DK = 128;
  const float inv_scale = 0.08838834764831845f;  // 1/sqrt(128)

  const float* x    = (const float*)d_in[0];
  // d_in[1] consciousness_state: truncated away by the reference's concat[:, :64]
  const float* wq   = (const float*)d_in[2];
  const float* wk   = (const float*)d_in[3];
  const float* wv   = (const float*)d_in[4];
  const float* wo   = (const float*)d_in[5];
  const float* wo_b = (const float*)d_in[6];
  const float* fw   = (const float*)d_in[7];
  const float* fbias= (const float*)d_in[8];
  const float* fm1w = (const float*)d_in[9];
  const float* fm1b = (const float*)d_in[10];
  const float* fm2w = (const float*)d_in[11];
  const float* fm2b = (const float*)d_in[12];
  const float* cmod = (const float*)d_in[13];
  float* out = (float*)d_out;

  size_t off = 0;
  auto alloc = [&](size_t bytes) -> void* {
    void* p = (char*)d_ws + off;
    off += (bytes + 255) & ~(size_t)255;
    return p;
  };
  unsigned short* xb    = (unsigned short*)alloc((size_t)S * D * 2);
  unsigned short* wqb   = (unsigned short*)alloc((size_t)D * D * 2);
  unsigned short* wkb   = (unsigned short*)alloc((size_t)D * D * 2);
  unsigned short* wvb   = (unsigned short*)alloc((size_t)D * D * 2);
  unsigned short* wob   = (unsigned short*)alloc((size_t)D * D * 2);
  unsigned short* fwb   = (unsigned short*)alloc((size_t)F * D * 2);
  unsigned short* qb    = (unsigned short*)alloc((size_t)S * D * 2);
  unsigned short* kb    = (unsigned short*)alloc((size_t)S * D * 2);
  unsigned short* vTb   = (unsigned short*)alloc((size_t)D * S * 2);  // transposed V
  float*          fp    = (float*)alloc((size_t)S * F * 4);
  unsigned short* fpb   = (unsigned short*)alloc((size_t)S * F * 2);
  unsigned short* gb    = (unsigned short*)alloc((size_t)S * F * 2);
  float*          mi    = (float*)alloc(F * 4);
  float*          raw   = (float*)alloc(F * 4);
  float*          metric= (float*)alloc(F * F * 4);
  float*          curv  = (float*)alloc(S * 4);
  float*          mod   = (float*)alloc((size_t)S * S * 4);
  float*          scores= (float*)alloc((size_t)S * S * 4);
  unsigned short* probs = (unsigned short*)alloc((size_t)S * S * 2);
  unsigned short* attnb = (unsigned short*)alloc((size_t)S * D * 2);
  float*          part  = (float*)alloc(S * 4);

  // 1) fp32 -> bf16 conversions for activations and weights
  launch_cvt(stream, x,  xb,  S * D);
  launch_cvt(stream, wq, wqb, D * D);
  launch_cvt(stream, wk, wkb, D * D);
  launch_cvt(stream, wv, wvb, D * D);
  launch_cvt(stream, wo, wob, D * D);
  launch_cvt(stream, fw, fwb, F * D);

  // 2) fiber projection: fp = x @ fw^T + fb   (fp32 out; N=64 -> small kernel)
  launch_gemm(stream, xb, D, fwb, D, fp, F, S, F, D, 1.0f,
              nullptr, 0, nullptr, 0.f, fbias, 0);
  launch_cvt(stream, fp, fpb, S * F);

  // 3) fiber metric path (tiny)
  col_mean_kernel<<<F, 256, 0, stream>>>(fp, mi, S, F);
  fiber_mlp_kernel<<<1, 64, 0, stream>>>(mi, fm1w, fm1b, fm2w, fm2b, raw);
  metric_kernel<<<16, 256, 0, stream>>>(raw, metric);
  fiber_g_kernel<<<S / 32, 64, 0, stream>>>(fp, metric, gb);
  curv_kernel<<<S / 256, 256, 0, stream>>>(fp, cmod, curv);

  // 4) q/k/v projections (bf16 out; V stored transposed feature-major)
  launch_gemm(stream, xb, D, wqb, D, qb, D, S, D, D, 1.0f,
              nullptr, 0, nullptr, 0.f, nullptr, GFLAG_OUT_BF16);
  launch_gemm(stream, xb, D, wkb, D, kb, D, S, D, D, 1.0f,
              nullptr, 0, nullptr, 0.f, nullptr, GFLAG_OUT_BF16);
  launch_gemm(stream, xb, D, wvb, D, vTb, S, S, D, D, 1.0f,
              nullptr, 0, nullptr, 0.f, nullptr, GFLAG_OUT_BF16 | GFLAG_TRANSPOSE);

  // 5) mod = (fp @ metric) @ fp^T  — head-independent, computed once (fp32)
  launch_gemm(stream, gb, F, fpb, F, mod, S, S, S, F, 1.0f,
              nullptr, 0, nullptr, 0.f, nullptr, 0);

  // 6) per-head attention: scores (+0.1*(mod+curv)) -> softmax -> @V
  for (int h = 0; h < H; ++h) {
    launch_gemm(stream, qb + h * DK, D, kb + h * DK, D, scores, S, S, S, DK,
                inv_scale, mod, S, curv, 0.1f, nullptr, 0);
    softmax_rows<<<S, 256, 0, stream>>>(scores, probs, S);
    launch_gemm(stream, probs, S, vTb + (size_t)h * DK * S, S,
                attnb + h * DK, D, S, DK, S, 1.0f,
                nullptr, 0, nullptr, 0.f, nullptr, GFLAG_OUT_BF16);
  }

  // 7) output projection: out = attn @ wo^T + wo_b  (fp32 -> d_out)
  launch_gemm(stream, attnb, D, wob, D, out, D, S, D, D, 1.0f,
              nullptr, 0, nullptr, 0.f, wo_b, 0);

  // 8) ci.mean() = mean(mod) + mean(curv)  (deterministic two-stage reduce)
  reduce_rows_kernel<<<S, 256, 0, stream>>>(mod, part, S);
  reduce_final_kernel<<<1, 256, 0, stream>>>(part, curv, out + (size_t)S * D, S);
}
